// PCLSTM_25778393711280
// MI455X (gfx1250) — compile-verified, matrix operand order checked by reading, not run on hardware
//
#include <hip/hip_runtime.h>
#include <hip/hip_bf16.h>

// ---------------------------------------------------------------------------
// LSTM forward: B=512, C=128, H=128, T=512, fp32 in/out.
// v3: all LDS fragment addresses expressed as loop-invariant pointers + small
// immediate offsets so the hot loop has no per-iteration address VALU (which
// was also triggering WMMA->VALU hazard v_nops). Hardware bf16 converts.
// MT=16 -> 32 workgroups; x prefetched one timestep ahead into VGPRs;
// bf16 weights resident in LDS; cell state in VGPRs across all 512 steps.
// ---------------------------------------------------------------------------

typedef __attribute__((ext_vector_type(16))) __bf16 v16bf;
typedef __attribute__((ext_vector_type(8)))  float  v8f;

#define BB   512
#define CC   128
#define HH   128
#define TT   512
#define KDIM 256            // C + H
#define MT   16             // batch rows per workgroup
#define SROW 264            // padded row length (bf16 elems) for S tile
#define WROW 264            // padded row length (bf16 elems) for weights
#define NTHREADS 256        // 8 waves
#define XPT  ((MT * CC) / NTHREADS)   // x elements prefetched per thread = 8

__device__ __forceinline__ unsigned short f2bf(float f) {
    union { __hip_bfloat16 h; unsigned short u; } cv;
    cv.h = __float2bfloat16(f);          // hardware cvt on gfx1250
    return cv.u;
}
__device__ __forceinline__ float sigmoid_fast(float x) {
    return 1.0f / (1.0f + __expf(-x));
}
__device__ __forceinline__ float tanh_fast(float x) {
    float e = __expf(2.0f * x);
    return (e - 1.0f) / (e + 1.0f);
}

union FragU { v16bf v; uint4 q[2]; };

__global__ void __launch_bounds__(NTHREADS, 1)
pclstm_wmma_kernel(const float* __restrict__ x,
                   const float* __restrict__ Wf, const float* __restrict__ bf_,
                   const float* __restrict__ Wi, const float* __restrict__ bi_,
                   const float* __restrict__ Wu, const float* __restrict__ bu_,
                   const float* __restrict__ Wo, const float* __restrict__ bo_,
                   float* __restrict__ out)
{
    extern __shared__ __align__(16) unsigned short smem[];
    unsigned short* sW = smem;                       // 4 * HH * WROW bf16
    unsigned short* sS = smem + 4 * HH * WROW;       // MT * SROW bf16 ([x_t, h])

    const int tid    = threadIdx.x;
    const int lane   = tid & 31;
    const int wave   = tid >> 5;          // 0..7, owns 16 hidden columns
    const int hbase  = wave * 16;
    const int nCol   = lane & 15;         // N index within 16x16 tile
    const int hiHalf = lane >> 4;         // 0: lanes 0-15, 1: lanes 16-31
    const int bbase  = blockIdx.x * MT;

    // ---- one-time: convert all four weight matrices to bf16, transposed into
    // LDS. Global W layout: (K=256, N=128) row-major; LDS: Wt[g][n][k].
    const float* Wg[4] = {Wf, Wi, Wu, Wo};
    for (int idx = tid; idx < 4 * HH * KDIM; idx += NTHREADS) {
        int g = idx >> 15;              // / (128*256)
        int r = idx & 32767;
        int n = r >> 8;
        int k = r & 255;
        sW[g * HH * WROW + n * WROW + k] = f2bf(Wg[g][k * HH + n]);
    }
    // ---- zero h region of S tile (h0 = 0)
    for (int idx = tid; idx < MT * HH; idx += NTHREADS) {
        int row = idx >> 7;
        int c   = idx & 127;
        sS[row * SROW + CC + c] = (unsigned short)0;
    }

    // ---- loop-invariant LDS fragment pointers (all ds_loads in the hot loop
    // become base + immediate offset; no per-iteration address VALU).
    const unsigned short* ap = sS + (lane & 15) * SROW + hiHalf * 8;
    const unsigned short* w0 = sW + 0 * HH * WROW + (hbase + nCol) * WROW + hiHalf * 8;
    const unsigned short* w1 = sW + 1 * HH * WROW + (hbase + nCol) * WROW + hiHalf * 8;
    const unsigned short* w2 = sW + 2 * HH * WROW + (hbase + nCol) * WROW + hiHalf * 8;
    const unsigned short* w3 = sW + 3 * HH * WROW + (hbase + nCol) * WROW + hiHalf * 8;
    unsigned short* hst = sS + (hiHalf * 8) * SROW + CC + hbase + nCol; // h store base (row r adds r*SROW)

    // per-lane gate biases for owned N column
    const float bF = bf_[hbase + nCol];
    const float bI = bi_[hbase + nCol];
    const float bU = bu_[hbase + nCol];
    const float bO = bo_[hbase + nCol];

    // per-lane global output base for each C/D row r: out[(b*H + h)*T]
    float* outp[8];
    #pragma unroll
    for (int r = 0; r < 8; ++r) {
        const int row = r + hiHalf * 8;
        outp[r] = out + ((size_t)(bbase + row) * HH + hbase + nCol) * TT;
    }

    // cell state in registers (C/D layout: row = r + hiHalf*8, col = nCol)
    float creg[8];
    #pragma unroll
    for (int r = 0; r < 8; ++r) creg[r] = 0.0f;

    const v8f vzero = {0.f, 0.f, 0.f, 0.f, 0.f, 0.f, 0.f, 0.f};

    // ---- prologue: prefetch x_0 into registers (8 values per thread)
    float xcur[XPT], xnext[XPT];
    #pragma unroll
    for (int i = 0; i < XPT; ++i) {
        int idx = tid + i * NTHREADS;
        int row = idx >> 7;
        int c   = idx & 127;
        xcur[i] = x[((size_t)(bbase + row) * CC + c) * TT + 0];
    }

    for (int t = 0; t < TT; ++t) {
        // ---- commit prefetched x_t tile into S (cols 0..127), bf16
        #pragma unroll
        for (int i = 0; i < XPT; ++i) {
            int idx = tid + i * NTHREADS;
            int row = idx >> 7;
            int c   = idx & 127;
            sS[row * SROW + c] = f2bf(xcur[i]);
        }
        // ---- issue next step's x loads; latency hidden behind this step
        if (t + 1 < TT) {
            #pragma unroll
            for (int i = 0; i < XPT; ++i) {
                int idx = tid + i * NTHREADS;
                int row = idx >> 7;
                int c   = idx & 127;
                xnext[i] = x[((size_t)(bbase + row) * CC + c) * TT + (t + 1)];
            }
        }
        __syncthreads();   // S (x part + h from prev step) ready for GEMM reads

        // ---- accumulators: one 16x16 f32 tile per gate
        v8f accF = vzero, accI = vzero, accU = vzero, accO = vzero;

        // ---- K loop: 8 steps of k=32, v_wmma_f32_16x16x32_bf16
        #pragma unroll
        for (int ks = 0; ks < 8; ++ks) {
            FragU a, b0, b1, b2, b3;
            a.q[0]  = *(const uint4*)(ap + ks * 32);
            a.q[1]  = *(const uint4*)(ap + ks * 32 + 16);
            b0.q[0] = *(const uint4*)(w0 + ks * 32);
            b0.q[1] = *(const uint4*)(w0 + ks * 32 + 16);
            b1.q[0] = *(const uint4*)(w1 + ks * 32);
            b1.q[1] = *(const uint4*)(w1 + ks * 32 + 16);
            b2.q[0] = *(const uint4*)(w2 + ks * 32);
            b2.q[1] = *(const uint4*)(w2 + ks * 32 + 16);
            b3.q[0] = *(const uint4*)(w3 + ks * 32);
            b3.q[1] = *(const uint4*)(w3 + ks * 32 + 16);
            accF = __builtin_amdgcn_wmma_f32_16x16x32_bf16(
                false, a.v, false, b0.v, (short)0, accF, false, false);
            accI = __builtin_amdgcn_wmma_f32_16x16x32_bf16(
                false, a.v, false, b1.v, (short)0, accI, false, false);
            accU = __builtin_amdgcn_wmma_f32_16x16x32_bf16(
                false, a.v, false, b2.v, (short)0, accU, false, false);
            accO = __builtin_amdgcn_wmma_f32_16x16x32_bf16(
                false, a.v, false, b3.v, (short)0, accO, false, false);
        }
        __syncthreads();   // GEMM reads done; safe to overwrite S below

        // ---- gates, cell update, write h (LDS bf16 + global fp32)
        #pragma unroll
        for (int r = 0; r < 8; ++r) {
            float fv = sigmoid_fast(accF[r] + bF);
            float iv = sigmoid_fast(accI[r] + bI);
            float uv = tanh_fast   (accU[r] + bU);
            float ov = sigmoid_fast(accO[r] + bO);
            float cv = creg[r] * fv + iv * uv;
            creg[r] = cv;
            float hv = ov * tanh_fast(cv);

            hst[r * SROW] = f2bf(hv);    // h for next step's GEMM (bf16)
            outp[r][t]    = hv;          // h_seq output (fp32)
        }

        // rotate prefetch buffer
        #pragma unroll
        for (int i = 0; i < XPT; ++i) xcur[i] = xnext[i];
    }

    // ---- final cell state: out[B*H*T + b*H + h]
    #pragma unroll
    for (int r = 0; r < 8; ++r) {
        const int row = r + hiHalf * 8;
        out[(size_t)BB * HH * TT + (size_t)(bbase + row) * HH + hbase + nCol]
            = creg[r];
    }
}

extern "C" void kernel_launch(void* const* d_in, const int* in_sizes, int n_in,
                              void* d_out, int out_size, void* d_ws, size_t ws_size,
                              hipStream_t stream) {
    const float* x  = (const float*)d_in[0];
    const float* Wf = (const float*)d_in[1];
    const float* bf = (const float*)d_in[2];
    const float* Wi = (const float*)d_in[3];
    const float* bi = (const float*)d_in[4];
    const float* Wu = (const float*)d_in[5];
    const float* bu = (const float*)d_in[6];
    const float* Wo = (const float*)d_in[7];
    const float* bo = (const float*)d_in[8];
    float* out = (float*)d_out;

    const size_t shmem = (size_t)(4 * HH * WROW + MT * SROW) * sizeof(unsigned short); // ~273 KB
    dim3 grid(BB / MT);   // 32 workgroups, one per 16 batch rows
    dim3 block(NTHREADS); // 8 waves
    pclstm_wmma_kernel<<<grid, block, shmem, stream>>>(
        x, Wf, bf, Wi, bi, Wu, bu, Wo, bo, out);
}
